// Attention_62311385530818
// MI455X (gfx1250) — matrix operand order repaired and checked by reading, not compiled, not run on hardware
//
#include <hip/hip_runtime.h>
#include <stdint.h>

// ---------------------------------------------------------------------------
// MHA forward for MI455X (gfx1250, wave32, WMMA).
// Compute-bound (~137 GFLOP vs ~100MB traffic) -> bf16 WMMA, f32 accumulate.
// Working set (~90MB bf16) fits in the 192MB L2. Attention stages K/V chunks
// into LDS with async global->LDS copies (ASYNCcnt), double-buffered, shared
// by the 4 waves of a block (4x less L2 read traffic than per-wave loads).
// ---------------------------------------------------------------------------

typedef __bf16 bf16;
typedef __bf16 v16bf __attribute__((ext_vector_type(16)));
typedef __bf16 v8bf  __attribute__((ext_vector_type(8)));
typedef float  v8f   __attribute__((ext_vector_type(8)));

#define DEV __device__ __forceinline__

static constexpr int BATCH  = 4;
static constexpr int SEQ    = 2048;
static constexpr int DMODEL = 1024;
static constexpr int HEADS  = 16;
static constexpr int DK     = 64;
static constexpr int MTOT   = BATCH * SEQ;         // 8192 rows

// ---- WMMA wrapper: D = A(16x32 bf16) * B(32x16 bf16) + C(16x16 f32) -------
DEV v8f wmma_bf16(v16bf a, v16bf b, v8f c) {
  return __builtin_amdgcn_wmma_f32_16x16x32_bf16(
      false, a, false, b, (short)0, c, false, false);
}

// ---- DPP xor-butterfly max within 16-lane halves (no LDS, no waits) -------
template <int CTRL>
DEV float dpp_movf(float x) {
  return __int_as_float(__builtin_amdgcn_update_dpp(
      0, __float_as_int(x), CTRL, 0xF, 0xF, true));
}
DEV float halfgroup_max(float x) {
  x = fmaxf(x, dpp_movf<0xB1>(x));    // quad_perm [1,0,3,2]  : xor 1
  x = fmaxf(x, dpp_movf<0x4E>(x));    // quad_perm [2,3,0,1]  : xor 2
  x = fmaxf(x, dpp_movf<0x164>(x));   // ROW_XMASK 4          : xor 4
  x = fmaxf(x, dpp_movf<0x168>(x));   // ROW_XMASK 8          : xor 8
  return x;
}

// ---- async global->LDS 16B copy (GLOBAL_LOAD_ASYNC_TO_LDS_B128) -----------
// Tracked by ASYNCcnt; LDS dest = 32-bit wave-relative LDS byte address
// (low 32 bits of the generic pointer), global src = 64-bit VGPR pair.
DEV void async_ld16(const bf16* lds_dst, const bf16* gsrc) {
  uint32_t lo = (uint32_t)(uintptr_t)lds_dst;
  asm volatile("global_load_async_to_lds_b128 %0, %1, off"
               :: "v"(lo), "v"(gsrc) : "memory");
}
DEV void wait_async0() {
  asm volatile("s_wait_asynccnt 0x0" ::: "memory");
}

// ---- A-matrix tile (16x32, MxK) from row-major [*, ld] ---------------------
// ISA 7.12.2: lane L -> row L&15; K = (L>>4)*8 + {0..7} and {16..23}.
DEV v16bf load_a16x32(const bf16* p, int ld) {
  const int lane = threadIdx.x & 31;
  const bf16* q = p + (size_t)(lane & 15) * ld + ((lane >> 4) << 3);
  v8bf lo = *(const v8bf*)(q);
  v8bf hi = *(const v8bf*)(q + 16);
  v16bf r;
#pragma unroll
  for (int i = 0; i < 8; ++i) { r[i] = lo[i]; r[i + 8] = hi[i]; }
  return r;
}

// ---- B-matrix tile (32x16, KxN): B[k][n] = src[n*ld + k] (src [N,K] rows) --
// ISA 7.12.2 (B 32x16): lane L -> col L&15; K = (L>>4)*8 + {0..7} and {16..23}.
DEV v16bf load_b32x16(const bf16* p, int ld) {
  const int lane = threadIdx.x & 31;
  const bf16* q = p + (size_t)(lane & 15) * ld + ((lane >> 4) << 3);
  v8bf lo = *(const v8bf*)(q);
  v8bf hi = *(const v8bf*)(q + 16);
  v16bf r;
#pragma unroll
  for (int i = 0; i < 8; ++i) { r[i] = lo[i]; r[i + 8] = hi[i]; }
  return r;
}

// ---------------------------------------------------------------------------
// f32 -> bf16 convert (grid-stride)
// ---------------------------------------------------------------------------
__global__ void cvt_f32_bf16(const float* __restrict__ src,
                             bf16* __restrict__ dst, int n) {
  for (int i = blockIdx.x * blockDim.x + threadIdx.x; i < n;
       i += gridDim.x * blockDim.x)
    dst[i] = (bf16)src[i];
}

// ---------------------------------------------------------------------------
// GEMM: Y[M,N] = A[M,K](bf16) * W[N,K]^T(bf16) + bias.
// One wave owns a 16x64 tile; K-loop double-buffered so global loads of the
// next K-step overlap the 4 WMMAs of the current one.
// MODE 0: bf16 [B,H,S,Dk] (Q,K)  MODE 1: bf16 [B,H,Dk,S] (V)  MODE 2: f32 [M,N]
// ---------------------------------------------------------------------------
template <int MODE>
__global__ __launch_bounds__(256, 1)
void gemm16x64(const bf16* __restrict__ A, const bf16* __restrict__ W,
               const float* __restrict__ bias, void* __restrict__ outp,
               int M, int N, int K) {
  const int wave = (blockIdx.x * blockDim.x + threadIdx.x) >> 5;
  const int ntg  = N >> 6;
  const int m0   = (wave / ntg) << 4;
  const int n0   = (wave % ntg) << 6;

  const bf16* arow = A + (size_t)m0 * K;
  const bf16* wrow = W + (size_t)n0 * K;

  v8f acc[4] = {};
  v16bf a = load_a16x32(arow, K);
  v16bf b[4];
#pragma unroll
  for (int j = 0; j < 4; ++j) b[j] = load_b32x16(wrow + (size_t)(16 * j) * K, K);

  for (int k0 = 32; k0 < K; k0 += 32) {
    __builtin_prefetch(arow + k0 + 64, 0, 1);            // global_prefetch_b8
    v16bf an = load_a16x32(arow + k0, K);
    v16bf bn[4];
#pragma unroll
    for (int j = 0; j < 4; ++j)
      bn[j] = load_b32x16(wrow + (size_t)(16 * j) * K + k0, K);
#pragma unroll
    for (int j = 0; j < 4; ++j) acc[j] = wmma_bf16(a, b[j], acc[j]);
    a = an;
#pragma unroll
    for (int j = 0; j < 4; ++j) b[j] = bn[j];
  }
#pragma unroll
  for (int j = 0; j < 4; ++j) acc[j] = wmma_bf16(a, b[j], acc[j]);

  // C layout: VGPR r -> row m0 + r + 8*(lane>>4); col n0 + 16j + (lane&15).
  const int lane = threadIdx.x & 31;
  const int ncol = lane & 15, half = lane >> 4;
#pragma unroll
  for (int j = 0; j < 4; ++j) {
    const int e = n0 + 16 * j + ncol;
    const float bv = bias[e];
#pragma unroll
    for (int r = 0; r < 8; ++r) {
      const int m = m0 + r + 8 * half;
      const float v = acc[j][r] + bv;
      if (MODE == 2) {
        ((float*)outp)[(size_t)m * N + e] = v;
      } else {
        const int bb = m >> 11, s = m & 2047;   // SEQ=2048
        const int h = e >> 6, dk = e & 63;      // DK=64
        bf16* o = (bf16*)outp;
        if (MODE == 0)
          o[(((size_t)(bb * HEADS + h) * SEQ + s) << 6) + dk] = (bf16)v;
        else
          o[(((size_t)(bb * HEADS + h) * DK + dk) << 11) + s] = (bf16)v;
      }
    }
  }
}

// ---------------------------------------------------------------------------
// Flash attention. Block = (b,h, 64-query group); 4 waves x 16-query tiles.
// K/V 32-key chunks staged into double-buffered LDS with async global->LDS
// copies, shared by all 4 waves. Row max via DPP butterflies; row sum via a
// P*ones WMMA; P transposed C->A layout through a per-wave LDS bounce.
// Q,K: [B,H,S,Dk] bf16.  V: [B,H,Dk,S] bf16.  Out: [B,S,D] bf16.
// ---------------------------------------------------------------------------
__global__ __launch_bounds__(128, 1)
void attn_kernel(const bf16* __restrict__ Q, const bf16* __restrict__ Kt,
                 const bf16* __restrict__ Vt, bf16* __restrict__ O) {
  __shared__ __align__(16) bf16 kbuf[2][32 * 64];   // [key][d]   2 x 4KB
  __shared__ __align__(16) bf16 vbuf[2][64 * 32];   // [d][key]   2 x 4KB
  __shared__ __align__(16) bf16 plds[4][16 * 32];   // per-wave P bounce

  const int tid  = threadIdx.x;             // 0..127
  const int wib  = tid >> 5;
  const int lane = tid & 31;
  const int ncol = lane & 15, half = lane >> 4;

  const int bh = blockIdx.x >> 5;           // b*16+h  (64)
  const int qg = blockIdx.x & 31;           // query group (32)
  const int q0 = qg * 64 + wib * 16;

  const bf16* Ktb = Kt + (size_t)bh * SEQ * DK;   // [key][d]
  const bf16* Vtb = Vt + (size_t)bh * DK * SEQ;   // [d][key]

  const bf16* qbase = Q + ((size_t)bh * SEQ + q0) * DK;
  const v16bf aq0 = load_a16x32(qbase, DK);        // d = 0..31
  const v16bf aq1 = load_a16x32(qbase + 32, DK);   // d = 32..63

  v16bf bones;
#pragma unroll
  for (int i = 0; i < 16; ++i) bones[i] = (bf16)1.0f;

  float mrow[8], lrow[8];
#pragma unroll
  for (int r = 0; r < 8; ++r) { mrow[r] = -3.0e30f; lrow[r] = 0.0f; }
  v8f oacc[4] = {};

  // ---- stage chunk [kc, kc+32) into LDS buffer `buf` (512 x 16B) ----
  auto stage = [&](int kc, int buf) {
    // K chunk: contiguous 4KB block (rows are Dk=64 contiguous)
    const bf16* kg = Ktb + (size_t)kc * DK;
#pragma unroll
    for (int u = 0; u < 2; ++u) {
      const int seg = tid * 2 + u;                       // 0..255
      async_ld16(&kbuf[buf][seg * 8], kg + seg * 8);
    }
    // V chunk: 64 rows (d) x 64B, row stride SEQ
#pragma unroll
    for (int u = 0; u < 2; ++u) {
      const int seg = tid * 2 + u;                       // 0..255
      const int row = seg >> 2, c = (seg & 3) * 8;
      async_ld16(&vbuf[buf][row * 32 + c],
                 Vtb + (size_t)row * SEQ + kc + c);
    }
  };

  stage(0, 0);                              // prologue: chunk 0 in flight

  const float scale = 0.125f;               // 1/sqrt(64)
  const int NCHUNK = SEQ / 32;              // 64
  for (int ci = 0; ci < NCHUNK; ++ci) {
    const int cur = ci & 1;
    wait_async0();                          // chunk ci landed in LDS
    __syncthreads();                        // all waves see it; prev reads done
    if (ci + 1 < NCHUNK) stage((ci + 1) * 32, cur ^ 1);  // overlap next copy

    // ---- S = Q K^T from LDS K chunk ----
    const bf16* kb = &kbuf[cur][0];
    v16bf bk0 = load_b32x16(kb, DK);
    v16bf bk1 = load_b32x16(kb + 32, DK);
    v16bf bk2 = load_b32x16(kb + 16 * DK, DK);
    v16bf bk3 = load_b32x16(kb + 16 * DK + 32, DK);
    v8f s0 = {}, s1 = {};
    s0 = wmma_bf16(aq0, bk0, s0);
    s1 = wmma_bf16(aq0, bk2, s1);
    s0 = wmma_bf16(aq1, bk1, s0);
    s1 = wmma_bf16(aq1, bk3, s1);

    // ---- online softmax, C layout (row = r + 8*half, col = key) ----
#pragma unroll
    for (int r = 0; r < 8; ++r) {
      const float x0 = s0[r] * scale, x1 = s1[r] * scale;
      const float t = halfgroup_max(fmaxf(x0, x1));
      const float mnew = fmaxf(mrow[r], t);
      const float corr = __expf(mrow[r] - mnew);
      mrow[r] = mnew;
      lrow[r] *= corr;
#pragma unroll
      for (int j = 0; j < 4; ++j) oacc[j][r] *= corr;
      const int qrow = r + 8 * half;
      plds[wib][qrow * 32 + ncol]      = (bf16)__expf(x0 - mnew);
      plds[wib][qrow * 32 + 16 + ncol] = (bf16)__expf(x1 - mnew);
    }

    // ---- P in A layout; row sums via P*ones WMMA; O += P*V ----
    const v16bf ap = load_a16x32(&plds[wib][0], 32);
    v8f psum = {};
    psum = wmma_bf16(ap, bones, psum);      // every col = row sum of P
#pragma unroll
    for (int r = 0; r < 8; ++r) lrow[r] += psum[r];
#pragma unroll
    for (int j = 0; j < 4; ++j) {
      oacc[j] = wmma_bf16(ap, load_b32x16(&vbuf[cur][j * 16 * 32], 32), oacc[j]);
    }
  }

  // ---- epilogue: normalize, scatter to [B,S,D] bf16 ----
  const int b = bh >> 4, h = bh & 15;
#pragma unroll
  for (int r = 0; r < 8; ++r) {
    const float inv = 1.0f / lrow[r];
    const int s = q0 + r + 8 * half;
#pragma unroll
    for (int j = 0; j < 4; ++j) {
      const int e = h * DK + j * 16 + ncol;
      O[((size_t)(b * SEQ + s)) * DMODEL + e] = (bf16)(oacc[j][r] * inv);
    }
  }
}

// ---------------------------------------------------------------------------
// Host launcher
// ---------------------------------------------------------------------------
extern "C" void kernel_launch(void* const* d_in, const int* in_sizes, int n_in,
                              void* d_out, int out_size, void* d_ws, size_t ws_size,
                              hipStream_t stream) {
  const float* x  = (const float*)d_in[0];
  const float* Wq = (const float*)d_in[1];
  const float* bq = (const float*)d_in[2];
  const float* Wk = (const float*)d_in[3];
  const float* bk = (const float*)d_in[4];
  const float* Wv = (const float*)d_in[5];
  const float* bv = (const float*)d_in[6];
  const float* Wo = (const float*)d_in[7];
  const float* bo = (const float*)d_in[8];

  const int NX = MTOT * DMODEL;
  const int NW = DMODEL * DMODEL;

  bf16* ws  = (bf16*)d_ws;
  bf16* xb  = ws;
  bf16* wqb = xb  + (size_t)NX;
  bf16* wkb = wqb + (size_t)NW;
  bf16* wvb = wkb + (size_t)NW;
  bf16* wob = wvb + (size_t)NW;
  bf16* Qw  = wob + (size_t)NW;         // [B,H,S,Dk]
  bf16* Kw  = Qw  + (size_t)NX;         // [B,H,S,Dk]
  bf16* Vw  = Kw  + (size_t)NX;         // [B,H,Dk,S]
  bf16* AO  = Vw  + (size_t)NX;         // [B,S,D]
  (void)ws_size; (void)n_in; (void)in_sizes; (void)out_size;

  cvt_f32_bf16<<<2048, 256, 0, stream>>>(x,  xb,  NX);
  cvt_f32_bf16<<<512,  256, 0, stream>>>(Wq, wqb, NW);
  cvt_f32_bf16<<<512,  256, 0, stream>>>(Wk, wkb, NW);
  cvt_f32_bf16<<<512,  256, 0, stream>>>(Wv, wvb, NW);
  cvt_f32_bf16<<<512,  256, 0, stream>>>(Wo, wob, NW);

  gemm16x64<0><<<1024, 256, 0, stream>>>(xb, wqb, bq, Qw, MTOT, DMODEL, DMODEL);
  gemm16x64<0><<<1024, 256, 0, stream>>>(xb, wkb, bk, Kw, MTOT, DMODEL, DMODEL);
  gemm16x64<1><<<1024, 256, 0, stream>>>(xb, wvb, bv, Vw, MTOT, DMODEL, DMODEL);

  attn_kernel<<<2048, 128, 0, stream>>>(Qw, Kw, Vw, AO);

  gemm16x64<2><<<1024, 256, 0, stream>>>(AO, wob, bo, d_out, MTOT, DMODEL, DMODEL);
}
